// VQVAE2_9749575762197
// MI455X (gfx1250) — matrix operand order repaired and checked by reading, not compiled, not run on hardware
//
#include <hip/hip_runtime.h>
#include <hip/hip_bf16.h>
#include <math.h>

typedef __attribute__((ext_vector_type(16))) _Float16 v16h;
typedef __attribute__((ext_vector_type(8)))  _Float16 v8h;
typedef __attribute__((ext_vector_type(8)))  float    v8f;
typedef __attribute__((ext_vector_type(4)))  float    f32x4;
typedef __attribute__((ext_vector_type(4)))  unsigned u32x4;
typedef __attribute__((ext_vector_type(8)))  int      i32x8;
typedef __attribute__((ext_vector_type(4)))  int      i32x4;

#define MODE_CONV  0
#define MODE_CONVT 1

#ifndef USE_TDM
#define USE_TDM 1
#endif

#if USE_TDM
// ---------------------------------------------------------------------------
// Tensor Data Mover: load a 64-row x 32-col f32 tile of the weight matrix
// (row stride = Ktot elements) into LDS. D# per CDNA5 ISA ch.8:
//   group0: count=1 | lds_addr | global_addr | type=2
//   group1: data_size=4B, tensor_dim0/1 = remaining extents (OOB -> zeros),
//           tile_dim0=32, tile_dim1=64, tensor_dim0_stride=Ktot
// Issued by one wave; completion via s_wait_tensorcnt before the barrier.
// Toolchain uses the 6-arg builtin: (g0, g1, g2, g3, g4, cpol).
// ---------------------------------------------------------------------------
__device__ __forceinline__ void tdm_load_A(const float* w, int mtile, int k0,
                                           int Ktot, int Cout, float* ldsdst)
{
    const unsigned lds_off =
        (unsigned)(unsigned long long)(void*)ldsdst;   // low 32b = LDS offset
    const unsigned long long gaddr =
        (unsigned long long)(const void*)(w + (size_t)mtile * Ktot + k0);
    u32x4 g0;
    g0[0] = 1u;                                        // count=1 (user D#)
    g0[1] = lds_off;                                   // lds_addr
    g0[2] = (unsigned)(gaddr & 0xFFFFFFFFull);         // global_addr[31:0]
    g0[3] = (unsigned)((gaddr >> 32) & 0x01FFFFFFull)  // global_addr[56:32]
          | (2u << 30);                                // type=2 ("image")
    const unsigned td0 = (unsigned)(Ktot - k0);        // k extent from tile
    const unsigned td1 = (unsigned)(Cout - mtile);     // row extent from tile
    i32x8 g1;
    g1[0] = (int)(2u << 16);                           // data_size=2 (4 bytes)
    g1[1] = (int)((td0 & 0xFFFFu) << 16);              // tensor_dim0[15:0]
    g1[2] = (int)((td0 >> 16) | ((td1 & 0xFFFFu) << 16)); // td0 hi | td1 lo
    g1[3] = (int)((td1 >> 16) | (32u << 16));          // td1 hi | tile_dim0=32
    g1[4] = 64;                                        // tile_dim1=64, dim2=0
    g1[5] = Ktot;                                      // dim0_stride[31:0]
    g1[6] = 0;                                         // dim0_stride hi
    g1[7] = 0;
    const i32x4 z4 = {0, 0, 0, 0};
    const i32x8 z8 = {0, 0, 0, 0, 0, 0, 0, 0};
    __builtin_amdgcn_tensor_load_to_lds(g0, g1, z4, z4, z8, 0);
}
#endif

// ---------------------------------------------------------------------------
// Implicit-GEMM convolution (normal + transposed), WMMA f16 -> f32.
// 128 threads (4 waves); block tile 64(M=Cout) x 64(N=B*Ho*Wo); K-step 32.
// Double-buffered LDS; A tile staged f32 (TDM for normal conv), B tile f16.
// ---------------------------------------------------------------------------
__global__ __launch_bounds__(128)
void conv_igemm(const float* __restrict__ in, const float* __restrict__ w,
                const float* __restrict__ bias, const float* __restrict__ res,
                float* __restrict__ out,
                int B, int Cin, int Hin, int Win,
                int Cout, int Hout, int Wout,
                int ksz, int stride, int pad,
                int mode, int relu_in, int act)
{
    __shared__ float    lAf[2][64 * 32];   // 16 KB (f32 weight tiles)
    __shared__ _Float16 lBh[2][64 * 32];   //  8 KB (f16 im2col tiles)

    const int t     = threadIdx.x;
    const int lane  = t & 31;
    const int wave  = t >> 5;
    const int mtile = blockIdx.y * 64;
    const int ntile = blockIdx.x * 64;
    const int HWo   = Hout * Wout;
    const int Ntot  = B * HWo;
    const int kk2   = ksz * ksz;
    const int Ktot  = Cin * kk2;

    // ---- per-thread staging geometry (constant across K loop) ----
    const int nrow = t >> 1;          // local n row 0..63 (B tile)
    const int ks   = (t & 1) * 16;    // k half: 0 or 16
    const int ng   = ntile + nrow;
    const bool nvalid = (ng < Ntot);
    int bb = 0, ho = 0, wo = 0;
    if (nvalid) {
        bb = ng / HWo;
        const int r = ng - bb * HWo;
        ho = r / Wout;
        wo = r - ho * Wout;
    }
    const int q = ksz - 1 - pad;
    const int hbase = (mode == MODE_CONV) ? (ho * stride - pad) : (ho - q);
    const int wbase = (mode == MODE_CONV) ? (wo * stride - pad) : (wo - q);
    const long inbase = (long)bb * Cin * Hin * Win;

    const int am  = t >> 1;           // A tile row (manual path)
    const int aks = (t & 1) * 16;
    const int aco = mtile + am;

    float breg[16];
    float areg[16];

    // ---- branchless im2col gather into registers ----
    auto stageB = [&](int k0) {
        const int kbase = k0 + ks;
        int ci = kbase / kk2;
        int r  = kbase - ci * kk2;
        int kh = r / ksz;
        int kw = r - kh * ksz;
#pragma unroll
        for (int u = 0; u < 16; ++u) {
            bool ok = nvalid && (kbase + u < Ktot);
            int hi, wi;
            if (mode == MODE_CONV) {
                hi = hbase + kh;
                wi = wbase + kw;
                ok = ok && hi >= 0 && hi < Hin && wi >= 0 && wi < Win;
            } else {
                const int hd = hbase + kh;
                const int wd = wbase + kw;
                ok = ok && hd >= 0 && wd >= 0 &&
                     (hd % stride) == 0 && (wd % stride) == 0;
                hi = hd / stride;
                wi = wd / stride;
                ok = ok && hi < Hin && wi < Win;
            }
            const long a = ok ? (inbase + ((long)ci * Hin + hi) * Win + wi) : 0;
            float v = in[a];
            v = ok ? v : 0.0f;
            if (relu_in) v = fmaxf(v, 0.0f);
            breg[u] = v;
            ++kw;
            if (kw == ksz) { kw = 0; ++kh; if (kh == ksz) { kh = 0; ++ci; } }
        }
    };
    auto storeB = [&](int buf) {
        v8h h0, h1;
#pragma unroll
        for (int j = 0; j < 8; ++j) {
            h0[j] = (_Float16)breg[j];
            h1[j] = (_Float16)breg[8 + j];
        }
        *(v8h*)&lBh[buf][nrow * 32 + ks]     = h0;
        *(v8h*)&lBh[buf][nrow * 32 + ks + 8] = h1;
    };

    // ---- A tile: TDM (normal conv) or manual register staging ----
    const bool use_tdm_a = (mode == MODE_CONV) && (USE_TDM != 0);
    auto stageA = [&](int k0) {   // manual path only
        const int kbase = k0 + aks;
        if (mode == MODE_CONV) {
            if (aco < Cout && kbase + 15 < Ktot) {
                const f32x4* p = (const f32x4*)(w + (size_t)aco * Ktot + kbase);
#pragma unroll
                for (int j = 0; j < 4; ++j) {
                    const f32x4 x = p[j];
                    areg[4 * j + 0] = x[0]; areg[4 * j + 1] = x[1];
                    areg[4 * j + 2] = x[2]; areg[4 * j + 3] = x[3];
                }
            } else {
#pragma unroll
                for (int u = 0; u < 16; ++u) {
                    const bool ok = (aco < Cout) && (kbase + u < Ktot);
                    const long a = ok ? ((long)aco * Ktot + kbase + u) : 0;
                    const float v = w[a];
                    areg[u] = ok ? v : 0.0f;
                }
            }
        } else {
            // ConvT: w'[co][ci][kh][kw] = w[ci][co][K-1-kh][K-1-kw]
            int ci = kbase / kk2;
            int r  = kbase - ci * kk2;
            int kh = r / ksz;
            int kw = r - kh * ksz;
#pragma unroll
            for (int u = 0; u < 16; ++u) {
                const bool ok = (aco < Cout) && (kbase + u < Ktot);
                const long a = ok
                    ? (((long)(ci * Cout + aco) * ksz + (ksz - 1 - kh)) * ksz +
                       (ksz - 1 - kw))
                    : 0;
                const float v = w[a];
                areg[u] = ok ? v : 0.0f;
                ++kw;
                if (kw == ksz) { kw = 0; ++kh; if (kh == ksz) { kh = 0; ++ci; } }
            }
        }
    };
    auto storeA = [&](int buf) {
#pragma unroll
        for (int j = 0; j < 4; ++j) {
            f32x4 x;
            x[0] = areg[4 * j + 0]; x[1] = areg[4 * j + 1];
            x[2] = areg[4 * j + 2]; x[3] = areg[4 * j + 3];
            *(f32x4*)&lAf[buf][am * 32 + aks + 4 * j] = x;
        }
    };
    auto issueA = [&](int k0, int buf) {
#if USE_TDM
        if (use_tdm_a) {
            if (wave == 0)
                tdm_load_A(w, mtile, k0, Ktot, Cout, &lAf[buf][0]);
            return;
        }
#endif
        stageA(k0);
    };
    auto commitA = [&](int buf) {
#if USE_TDM
        if (use_tdm_a) {
            if (wave == 0) __builtin_amdgcn_s_wait_tensorcnt(0);
            return;
        }
#endif
        storeA(buf);
    };

    // ---- accumulators ----
    v8f acc[2][2];
#pragma unroll
    for (int i = 0; i < 2; ++i)
#pragma unroll
        for (int j = 0; j < 2; ++j)
#pragma unroll
            for (int e = 0; e < 8; ++e) acc[i][j][e] = 0.0f;

    const int wm = (wave >> 1) * 32;
    const int wn = (wave & 1) * 32;
    const int lm = lane & 15;
    const int gg = lane >> 4;

    // ---- software-pipelined K loop, double-buffered LDS ----
    int buf = 0;
    issueA(0, 0);
    stageB(0);
    storeB(0);
    commitA(0);
    __syncthreads();

    for (int k0 = 0; k0 < Ktot; k0 += 32) {
        const bool more = (k0 + 32) < Ktot;
        if (more) {
            issueA(k0 + 32, buf ^ 1);
            stageB(k0 + 32);
        }

        // fragments: A from f32 LDS (cvt at fragment time), B from f16 LDS
        v16h afrag[2], bfrag[2];
#pragma unroll
        for (int i = 0; i < 2; ++i) {
            const float* bp = &lAf[buf][(wm + i * 16 + lm) * 32];
            const f32x4 x0 = *(const f32x4*)(bp + gg * 8);
            const f32x4 x1 = *(const f32x4*)(bp + gg * 8 + 4);
            const f32x4 x2 = *(const f32x4*)(bp + 16 + gg * 8);
            const f32x4 x3 = *(const f32x4*)(bp + 16 + gg * 8 + 4);
#pragma unroll
            for (int j = 0; j < 4; ++j) {
                afrag[i][j]      = (_Float16)x0[j];
                afrag[i][4 + j]  = (_Float16)x1[j];
                afrag[i][8 + j]  = (_Float16)x2[j];
                afrag[i][12 + j] = (_Float16)x3[j];
            }
        }
#pragma unroll
        for (int i = 0; i < 2; ++i) {
            const v8h b0 = *(const v8h*)&lBh[buf][(wn + i * 16 + lm) * 32 + gg * 8];
            const v8h b1 = *(const v8h*)&lBh[buf][(wn + i * 16 + lm) * 32 + 16 + gg * 8];
#pragma unroll
            for (int j = 0; j < 8; ++j) { bfrag[i][j] = b0[j]; bfrag[i][8 + j] = b1[j]; }
        }

#pragma unroll
        for (int i = 0; i < 2; ++i)
#pragma unroll
            for (int j = 0; j < 2; ++j)
                acc[i][j] = __builtin_amdgcn_wmma_f32_16x16x32_f16(
                    false, afrag[i], false, bfrag[j], (short)0, acc[i][j],
                    false, false);

        if (more) {
            storeB(buf ^ 1);
            commitA(buf ^ 1);
        }
        __syncthreads();
        buf ^= 1;
    }

    // ---- epilogue: bias (+residual) (+sigmoid); n-decode hoisted ----
#pragma unroll
    for (int j = 0; j < 2; ++j) {
        const int n = ntile + wn + j * 16 + lm;
        if (n < Ntot) {
            const int b2  = n / HWo;
            const int r2  = n - b2 * HWo;
            const int ho2 = r2 / Wout;
            const int wo2 = r2 - ho2 * Wout;
            const long nb = (long)b2 * Cout * HWo + (long)ho2 * Wout + wo2;
#pragma unroll
            for (int i = 0; i < 2; ++i)
#pragma unroll
                for (int e = 0; e < 8; ++e) {
                    const int co = mtile + wm + i * 16 + gg * 8 + e;
                    if (co < Cout) {
                        const long idx = nb + (long)co * HWo;
                        float v = acc[i][j][e] + bias[co];
                        if (res) v += res[idx];
                        if (act == 1) v = 1.0f / (1.0f + expf(-v));
                        out[idx] = v;
                    }
                }
        }
    }
}

// ---------------------------------------------------------------------------
// GroupNorm (G=8): stats (block per (b,g)) + fused normalize/ReLU.
// ---------------------------------------------------------------------------
__global__ __launch_bounds__(256)
void gn_stats(const float* __restrict__ x, float* __restrict__ mean,
              float* __restrict__ rstd, int C, int HW)
{
    __shared__ float ss[256], ss2[256];
    const int bg = blockIdx.x;
    const int b  = bg >> 3;
    const int g  = bg & 7;
    const int Cg = C >> 3;
    const long base = ((long)b * C + (long)g * Cg) * HW;
    const long n    = (long)Cg * HW;
    float s = 0.0f, s2 = 0.0f;
    for (long i = threadIdx.x; i < n; i += 256) {
        const float v = x[base + i];
        s += v; s2 += v * v;
    }
    ss[threadIdx.x] = s; ss2[threadIdx.x] = s2;
    __syncthreads();
    for (int o = 128; o > 0; o >>= 1) {
        if ((int)threadIdx.x < o) {
            ss[threadIdx.x]  += ss[threadIdx.x + o];
            ss2[threadIdx.x] += ss2[threadIdx.x + o];
        }
        __syncthreads();
    }
    if (threadIdx.x == 0) {
        const float mu  = ss[0] / (float)n;
        const float var = ss2[0] / (float)n - mu * mu;
        mean[bg] = mu;
        rstd[bg] = rsqrtf(var + 1e-5f);
    }
}

__global__ __launch_bounds__(256)
void gn_apply(const float* __restrict__ x, float* __restrict__ y,
              const float* __restrict__ mean, const float* __restrict__ rstd,
              const float* __restrict__ gamma, const float* __restrict__ beta,
              int C, int HW, int relu)
{
    const long i = (long)blockIdx.x * 256 + threadIdx.x;
    const long total = 8L * C * HW;
    if (i >= total) return;
    const int c = (int)((i / HW) % C);
    const int b = (int)(i / ((long)C * HW));
    const int g = c / (C >> 3);
    float v = (x[i] - mean[b * 8 + g]) * rstd[b * 8 + g] * gamma[c] + beta[c];
    if (relu) v = fmaxf(v, 0.0f);
    y[i] = v;
}

// ---------------------------------------------------------------------------
// Channel-offset copy (concat / h_b extraction), in-place ReLU, zero.
// ---------------------------------------------------------------------------
__global__ __launch_bounds__(256)
void copy_ch(const float* __restrict__ src, float* __restrict__ dst,
             int Cs, int HW, int Cd, int coff, int relu)
{
    const long i = (long)blockIdx.x * 256 + threadIdx.x;
    const long total = 8L * Cs * HW;
    if (i >= total) return;
    const int  b = (int)(i / ((long)Cs * HW));
    const long r = i - (long)b * Cs * HW;
    float v = src[i];
    if (relu) v = fmaxf(v, 0.0f);
    dst[((long)b * Cd + coff) * HW + r] = v;
}

__global__ __launch_bounds__(256)
void relu_ip(float* __restrict__ x, long n)
{
    const long i = (long)blockIdx.x * 256 + threadIdx.x;
    if (i < n) x[i] = fmaxf(x[i], 0.0f);
}

__global__ __launch_bounds__(256)
void zero_f(float* __restrict__ x, long n)
{
    const long i = (long)blockIdx.x * 256 + threadIdx.x;
    if (i < n) x[i] = 0.0f;
}

// ---------------------------------------------------------------------------
// VQ: thread-per-position argmin over K=512 codes (D=64), codebook tiled in
// LDS. Fuses quantized gather, SSE (commitment loss), histogram, index write.
// ---------------------------------------------------------------------------
__global__ __launch_bounds__(256)
void vq_kernel(const float* __restrict__ z, const float* __restrict__ emb,
               float* __restrict__ q, float* __restrict__ idx_out,
               float* __restrict__ counts, float* __restrict__ sse,
               int H, int W)
{
    __shared__ float le[128 * 64]; // 32 KB codebook tile
    const int HW = H * W;
    const int N  = 8 * HW;
    const int n  = blockIdx.x * 256 + threadIdx.x;
    const bool active = (n < N);

    int b = 0, h = 0, wo = 0;
    float zv[64];
    float znorm = 0.0f;
    if (active) {
        b = n / HW;
        const int r = n - b * HW;
        h  = r / W;
        wo = r - h * W;
#pragma unroll
        for (int d = 0; d < 64; ++d) {
            const float v = z[((b * 64 + d) * H + h) * W + wo];
            zv[d] = v;
            znorm += v * v;
        }
    }

    float best = 3.4e38f;
    int bestk = 0;
    for (int t0 = 0; t0 < 512; t0 += 128) {
        for (int i = threadIdx.x; i < 128 * 64; i += 256)
            le[i] = emb[t0 * 64 + i];
        __syncthreads();
        if (active) {
            for (int c = 0; c < 128; ++c) {
                const float* e = &le[c * 64];
                float dot = 0.0f, en = 0.0f;
#pragma unroll
                for (int d = 0; d < 64; ++d) {
                    dot += zv[d] * e[d];
                    en  += e[d] * e[d];
                }
                const float dist = znorm - 2.0f * dot + en;
                if (dist < best) { best = dist; bestk = t0 + c; }
            }
        }
        __syncthreads();
    }

    if (active) {
        idx_out[n] = (float)bestk;
        float local = 0.0f;
#pragma unroll
        for (int d = 0; d < 64; ++d) {
            const float e = emb[bestk * 64 + d];
            q[((b * 64 + d) * H + h) * W + wo] = e;
            const float df = e - zv[d];
            local += df * df;
        }
        atomicAdd(sse, local);
        atomicAdd(&counts[bestk], 1.0f);
    }
}

__global__ void vq_finalize(const float* __restrict__ counts_t,
                            const float* __restrict__ counts_b,
                            const float* __restrict__ sse,
                            float* __restrict__ out_loss,
                            float* __restrict__ out_ppl,
                            float Nt, float Nb)
{
    if (threadIdx.x == 0 && blockIdx.x == 0) {
        float st = 0.0f, sb = 0.0f;
        for (int k = 0; k < 512; ++k) {
            const float pt = counts_t[k] / Nt;
            const float pb = counts_b[k] / Nb;
            st += pt * logf(pt + 1e-10f);
            sb += pb * logf(pb + 1e-10f);
        }
        *out_ppl  = 0.5f * (expf(-st) + expf(-sb));
        *out_loss = 0.25f * (sse[0] / (Nt * 64.0f) + sse[1] / (Nb * 64.0f));
    }
}

// ---------------------------------------------------------------------------
// Host orchestration
// ---------------------------------------------------------------------------
extern "C" void kernel_launch(void* const* d_in, const int* in_sizes, int n_in,
                              void* d_out, int out_size, void* d_ws,
                              size_t ws_size, hipStream_t stream)
{
    (void)in_sizes; (void)n_in; (void)out_size; (void)ws_size;
#define W(i) ((const float*)d_in[(i)])

    // ---- workspace layout (floats) ----
    float* ws = (float*)d_ws;
    size_t off = 0;
    auto alloc = [&](size_t n) { float* p = ws + off; off += n; return p; };
    float* buf0  = alloc(33554432); // 8x256x128x128
    float* buf1  = alloc(8388608);  // 8x256x64x64
    float* buf2  = alloc(8388608);
    float* buf3  = alloc(8388608);  // h_b (relu'd)
    float* buf4  = alloc(2097152);  // 8x256x32x32
    float* buf5  = alloc(2097152);
    float* buf6  = alloc(524288);   // 8x64x32x32  (h_t)
    float* buf7  = alloc(524288);   // q_t
    float* buf9  = alloc(16777216); // 8x512x64x64 fused
    float* buf10 = alloc(2097152);  // 8x64x64x64
    float* buf11 = alloc(2097152);
    float* buf12 = alloc(2097152);  // q_b
    float* gmean = alloc(64);
    float* grstd = alloc(64);
    float* counts_t = alloc(512);
    float* counts_b = alloc(512);
    float* sse      = alloc(2);

    // ---- d_out layout: x_recon | loss | ppl | idx_t | idx_b ----
    float* out_x     = (float*)d_out;
    float* out_loss  = out_x + 8L * 3 * 256 * 256;
    float* out_ppl   = out_loss + 1;
    float* out_idx_t = out_ppl + 1;             // 8*32*32
    float* out_idx_b = out_idx_t + 8 * 32 * 32; // 8*64*64

    auto conv = [&](const float* in, int widx, const float* res, float* out,
                    int Cin, int Hin, int Win, int Cout, int Hout, int Wout,
                    int ksz, int s, int p, int mode, int relu_in, int act) {
        const int Ntot = 8 * Hout * Wout;
        dim3 grid((Ntot + 63) / 64, (Cout + 63) / 64);
        conv_igemm<<<grid, 128, 0, stream>>>(in, W(widx), W(widx + 1), res, out,
                                             8, Cin, Hin, Win, Cout, Hout, Wout,
                                             ksz, s, p, mode, relu_in, act);
    };
    auto gnorm = [&](float* x, int gidx, int C, int HW) {
        gn_stats<<<64, 256, 0, stream>>>(x, gmean, grstd, C, HW);
        const long total = 8L * C * HW;
        gn_apply<<<(int)((total + 255) / 256), 256, 0, stream>>>(
            x, x, gmean, grstd, W(gidx), W(gidx + 1), C, HW, 1);
    };
    auto res_stack = [&](float* xbuf, float* tbuf, int widx0, int C, int H,
                         int Wd) {
        for (int i = 0; i < 3; ++i) {
            const int wi = widx0 + i * 4;
            conv(xbuf, wi, nullptr, tbuf, C, H, Wd, C, H, Wd, 3, 1, 1,
                 MODE_CONV, 1, 0);
            conv(tbuf, wi + 2, xbuf, xbuf, C, H, Wd, C, H, Wd, 1, 1, 0,
                 MODE_CONV, 1, 0);
        }
    };
    auto relu_inplace = [&](float* x, long n) {
        relu_ip<<<(int)((n + 255) / 256), 256, 0, stream>>>(x, n);
    };

    // zero accumulators (counts_t, counts_b, sse contiguous: 1026 floats)
    zero_f<<<5, 256, 0, stream>>>(counts_t, 1026);

    // ===== encoder_b: 256 -> 64 spatial =====
    conv(W(0), 1, nullptr, buf0, 3, 256, 256, 256, 128, 128, 4, 2, 1,
         MODE_CONV, 0, 0);                                    // eb_c1
    gnorm(buf0, 3, 256, 128 * 128);                           // eb_n1 + relu
    conv(buf0, 5, nullptr, buf1, 256, 128, 128, 256, 64, 64, 4, 2, 1,
         MODE_CONV, 0, 0);                                    // eb_c2
    gnorm(buf1, 7, 256, 64 * 64);                             // eb_n2 + relu
    conv(buf1, 9, nullptr, buf2, 256, 64, 64, 256, 64, 64, 3, 1, 1,
         MODE_CONV, 0, 0);                                    // eb_c3
    res_stack(buf2, buf1, 11, 256, 64, 64);                   // eb_res
    copy_ch<<<(int)((8L * 256 * 4096 + 255) / 256), 256, 0, stream>>>(
        buf2, buf3, 256, 4096, 256, 0, 1);                    // h_b = relu(x)

    // ===== encoder_t: 64 -> 32 spatial =====
    conv(buf3, 23, nullptr, buf4, 256, 64, 64, 256, 32, 32, 4, 2, 1,
         MODE_CONV, 0, 0);                                    // et_c1
    gnorm(buf4, 25, 256, 32 * 32);                            // et_n1 + relu
    conv(buf4, 27, nullptr, buf5, 256, 32, 32, 256, 32, 32, 3, 1, 1,
         MODE_CONV, 0, 0);                                    // et_c2
    res_stack(buf5, buf4, 29, 256, 32, 32);                   // et_res
    relu_inplace(buf5, 8L * 256 * 1024);
    conv(buf5, 41, nullptr, buf6, 256, 32, 32, 64, 32, 32, 1, 1, 0,
         MODE_CONV, 0, 0);                                    // et_c3 -> h_t

    // ===== VQ top (emb_t = d_in[43]) =====
    vq_kernel<<<(8 * 32 * 32 + 255) / 256, 256, 0, stream>>>(
        buf6, W(43), buf7, out_idx_t, counts_t, sse + 0, 32, 32);

    // ===== top decoder: 32 -> 64 spatial =====
    conv(buf7, 44, nullptr, buf4, 64, 32, 32, 256, 32, 32, 3, 1, 1,
         MODE_CONV, 0, 0);                                    // td_c1
    res_stack(buf4, buf5, 46, 256, 32, 32);                   // td_res
    relu_inplace(buf4, 8L * 256 * 1024);
    conv(buf4, 58, nullptr, buf2, 256, 32, 32, 256, 64, 64, 4, 2, 1,
         MODE_CONVT, 0, 0);                                   // td_up
    gnorm(buf2, 60, 256, 64 * 64);                            // td_n + relu

    // ===== concat [h_b, up_t] -> 512ch, bottom pre-quant =====
    copy_ch<<<(int)((8L * 256 * 4096 + 255) / 256), 256, 0, stream>>>(
        buf3, buf9, 256, 4096, 512, 0, 0);
    copy_ch<<<(int)((8L * 256 * 4096 + 255) / 256), 256, 0, stream>>>(
        buf2, buf9, 256, 4096, 512, 256, 0);
    gnorm(buf9, 62, 512, 64 * 64);                            // bp_n0 + relu
    conv(buf9, 64, nullptr, buf10, 512, 64, 64, 64, 64, 64, 3, 1, 1,
         MODE_CONV, 0, 0);                                    // bp_c1
    res_stack(buf10, buf11, 66, 64, 64, 64);                  // bp_res
    relu_inplace(buf10, 8L * 64 * 4096);

    // ===== VQ bottom (emb_b = d_in[78]) =====
    vq_kernel<<<(8 * 64 * 64 + 255) / 256, 256, 0, stream>>>(
        buf10, W(78), buf12, out_idx_b, counts_b, sse + 1, 64, 64);

    // ===== decoder: 64 -> 256 spatial =====
    conv(buf12, 79, nullptr, buf1, 64, 64, 64, 256, 64, 64, 3, 1, 1,
         MODE_CONV, 0, 0);                                    // dec_c1
    res_stack(buf1, buf2, 81, 256, 64, 64);                   // dec_res
    relu_inplace(buf1, 8L * 256 * 4096);
    conv(buf1, 93, nullptr, buf0, 256, 64, 64, 256, 128, 128, 4, 2, 1,
         MODE_CONVT, 0, 0);                                   // dec_up1
    gnorm(buf0, 95, 256, 128 * 128);                          // dec_n1 + relu
    conv(buf0, 97, nullptr, out_x, 256, 128, 128, 3, 256, 256, 4, 2, 1,
         MODE_CONVT, 0, 1);                                   // dec_up2+sigmoid

    vq_finalize<<<1, 32, 0, stream>>>(counts_t, counts_b, sse, out_loss,
                                      out_ppl, 8192.0f, 32768.0f);
#undef W
}
